// MaskedAttentionV1_39556648796259
// MI455X (gfx1250) — matrix operand order repaired and checked by reading, not compiled, not run on hardware
//
#include <hip/hip_runtime.h>

typedef __attribute__((ext_vector_type(16))) __bf16 bf16x16;
typedef __attribute__((ext_vector_type(8)))  __bf16 bf16x8;
typedef __attribute__((ext_vector_type(8)))  float  f32x8;

static constexpr int S_SEQ = 8192;
static constexpr int D_DIM = 1024;
static constexpr int BM = 128, BN = 128, BK = 32;
static constexpr int LDT = 40;   // padded LDS row stride (bf16) = 80B -> conflict-free frag reads

// ---------------- fp32 -> bf16, vectorized (float4 -> 4x bf16) ----------------
__global__ void cvt_f32_bf16_v4(const float4* __restrict__ in, uint2* __restrict__ out, int n4) {
  int i = blockIdx.x * blockDim.x + threadIdx.x;
  if (i < n4) {
    float4 f = in[i];
    union { __bf16 e[4]; uint2 u; } o;
    o.e[0] = (__bf16)f.x; o.e[1] = (__bf16)f.y; o.e[2] = (__bf16)f.z; o.e[3] = (__bf16)f.w;
    out[i] = o.u;
  }
}

// ---------------- fp32 [R][C] -> bf16 [C][R] tiled transpose-convert ----------------
__global__ __launch_bounds__(256) void cvt_transpose_f32_bf16(
    const float* __restrict__ in, __bf16* __restrict__ out, int R, int C) {
  __shared__ float t[32][33];
  const int c0 = blockIdx.x * 32, r0 = blockIdx.y * 32;
  const int tx = threadIdx.x & 31, ty = threadIdx.x >> 5;  // 32 x 8
#pragma unroll
  for (int i = 0; i < 4; ++i)
    t[ty + i * 8][tx] = in[(size_t)(r0 + ty + i * 8) * C + c0 + tx];
  __syncthreads();
#pragma unroll
  for (int i = 0; i < 4; ++i)
    out[(size_t)(c0 + ty + i * 8) * R + r0 + tx] = (__bf16)t[tx][ty + i * 8];
}

// wait for this wave's outstanding async global->LDS copies
__device__ __forceinline__ void async_wait0() {
  asm volatile("s_wait_asynccnt 0x0" ::: "memory");
}

// ---------------- tiled WMMA bf16 GEMM (A @ B^T form) ----------------
// Double-buffered via GLOBAL_LOAD_ASYNC_TO_LDS_B128: global tiles stream straight
// into LDS (ASYNCcnt), zero staging VGPRs, overlapped with the WMMA block.
// C[M,N] = scale * A[M,K] @ B^T   with A row-major [M][K], B row-major [N][K]
template<bool OUT_BF16, bool OUT_TRANS, bool CAUSAL_SKIP, bool CAUSAL_KLIM>
__global__ __launch_bounds__(256) void gemm_wmma_bf16(
    const __bf16* __restrict__ A, const __bf16* __restrict__ B,
    void* __restrict__ Cout, int M, int N, int K, float scale)
{
  __shared__ __align__(16) __bf16 As[2][BM * LDT];
  __shared__ __align__(16) __bf16 Bs[2][BN * LDT];

  const int m0 = blockIdx.y * BM;
  const int n0 = blockIdx.x * BN;
  if (CAUSAL_SKIP && n0 >= m0 + BM) return;   // strictly-upper block: never read by softmax

  const int tid  = threadIdx.x;
  const int lane = tid & 31;
  const int wave = tid >> 5;      // 8 waves (wave32)
  const int wm   = wave >> 1;     // 0..3 : 32 output rows each
  const int wn   = wave & 1;      // 0..1 : 64 output cols each
  const int r    = lane & 15;
  const int h    = lane >> 4;

  f32x8 acc[2][4] = {};

  int kend = K;
  if (CAUSAL_KLIM) { int ke = m0 + BM; kend = ke < K ? ke : K; }  // multiple of BK
  const int kSteps = kend / BK;

  // per-thread staging coordinates: two 16B chunks of a BM x BK tile
  const int r0c = (tid * 2) >> 2,     c0c = ((tid * 2) & 3) * 8;
  const int r1c = (tid * 2 + 1) >> 2, c1c = ((tid * 2 + 1) & 3) * 8;

  // loop-invariant per-thread global base pointers (only uniform k0 advances)
  const __bf16* pa0 = A + (size_t)(m0 + r0c) * K + c0c;
  const __bf16* pa1 = A + (size_t)(m0 + r1c) * K + c1c;
  const __bf16* pb0 = B + (size_t)(n0 + r0c) * K + c0c;
  const __bf16* pb1 = B + (size_t)(n0 + r1c) * K + c1c;

  // per-thread LDS destinations, both buffers (low 32 bits of a generic
  // pointer to __shared__ == wave-relative LDS byte address)
  unsigned lA[2], lA1[2], lB[2], lB1[2];
#pragma unroll
  for (int b = 0; b < 2; ++b) {
    lA [b] = (unsigned)(uintptr_t)&As[b][r0c * LDT + c0c];
    lA1[b] = (unsigned)(uintptr_t)&As[b][r1c * LDT + c1c];
    lB [b] = (unsigned)(uintptr_t)&Bs[b][r0c * LDT + c0c];
    lB1[b] = (unsigned)(uintptr_t)&Bs[b][r1c * LDT + c1c];
  }

  auto stage_async = [&](int b, int k0) {
    asm volatile("global_load_async_to_lds_b128 %0, %1, off"
                 :: "v"(lA [b]), "v"(pa0 + k0) : "memory");
    asm volatile("global_load_async_to_lds_b128 %0, %1, off"
                 :: "v"(lA1[b]), "v"(pa1 + k0) : "memory");
    asm volatile("global_load_async_to_lds_b128 %0, %1, off"
                 :: "v"(lB [b]), "v"(pb0 + k0) : "memory");
    asm volatile("global_load_async_to_lds_b128 %0, %1, off"
                 :: "v"(lB1[b]), "v"(pb1 + k0) : "memory");
  };

  stage_async(0, 0);
  async_wait0();
  __syncthreads();

  for (int s = 0; s < kSteps; ++s) {
    const int cur = s & 1;
    if (s + 1 < kSteps) stage_async(cur ^ 1, (s + 1) * BK);   // DMA next tile -> other buffer
    if (s + 2 < kSteps) {  // warm L2 for the tile after next (global_prefetch_b8)
      __builtin_prefetch(pa0 + (s + 2) * BK, 0, 1);
      __builtin_prefetch(pb0 + (s + 2) * BK, 0, 1);
    }

    // ---- fragments per ISA VGPR layouts ----
    // A 16x32: lane(r,h) holds K = {8h..8h+7} then {16+8h..16+8h+7} of row r
    bf16x16 af[2];
#pragma unroll
    for (int fm = 0; fm < 2; ++fm) {
      const __bf16* p = &As[cur][(wm * 32 + fm * 16 + r) * LDT + h * 8];
      union { bf16x16 v; bf16x8 q[2]; } u;
      u.q[0] = *reinterpret_cast<const bf16x8*>(p);
      u.q[1] = *reinterpret_cast<const bf16x8*>(p + 16);
      af[fm] = u.v;
    }
    // B 32x16: lane(r,h) holds K = {8h..8h+7} then {16+8h..16+8h+7} of column r
    bf16x16 bfr[4];
#pragma unroll
    for (int fn = 0; fn < 4; ++fn) {
      const __bf16* p = &Bs[cur][(wn * 64 + fn * 16 + r) * LDT + h * 8];
      union { bf16x16 v; bf16x8 q[2]; } u;
      u.q[0] = *reinterpret_cast<const bf16x8*>(p);
      u.q[1] = *reinterpret_cast<const bf16x8*>(p + 16);
      bfr[fn] = u.v;
    }
#pragma unroll
    for (int fm = 0; fm < 2; ++fm)
#pragma unroll
      for (int fn = 0; fn < 4; ++fn)
        acc[fm][fn] = __builtin_amdgcn_wmma_f32_16x16x32_bf16(
            false, af[fm], false, bfr[fn], (short)0, acc[fm][fn], false, false);

    async_wait0();     // my async writes to the other buffer have landed in LDS
    __syncthreads();   // everyone's have
  }

  // ---- epilogue: C/D fragment element = (M = j + 8h, N = r) ----
#pragma unroll
  for (int fm = 0; fm < 2; ++fm)
#pragma unroll
    for (int fn = 0; fn < 4; ++fn) {
      const int rowb = m0 + wm * 32 + fm * 16 + h * 8;
      const int col  = n0 + wn * 64 + fn * 16 + r;
      if constexpr (OUT_TRANS) {
        // [N][M] layout: the 8 fragment rows are consecutive -> one 16B store
        union { bf16x8 v; uint4 u; } o;
#pragma unroll
        for (int j = 0; j < 8; ++j) o.v[j] = (__bf16)(acc[fm][fn][j] * scale);
        *reinterpret_cast<uint4*>(reinterpret_cast<__bf16*>(Cout) + (size_t)col * M + rowb) = o.u;
      } else {
#pragma unroll
        for (int j = 0; j < 8; ++j) {
          float val = acc[fm][fn][j] * scale;
          if constexpr (OUT_BF16)
            reinterpret_cast<__bf16*>(Cout)[(size_t)(rowb + j) * N + col] = (__bf16)val;
          else
            reinterpret_cast<float*>(Cout)[(size_t)(rowb + j) * N + col] = val;
        }
      }
    }
}

// ---------------- causal softmax: one block per row, row staged in LDS ----------------
__global__ __launch_bounds__(256) void softmax_causal(
    const float* __restrict__ Sc, __bf16* __restrict__ P, int n)
{
  __shared__ float rowbuf[S_SEQ];   // 32 KB: full score row (<= 320 KB/WGP)
  __shared__ float red[256];
  const int i   = blockIdx.x;
  const int tid = threadIdx.x;
  const int len  = i + 1;
  const int jmax = ((i >> 7) + 1) << 7;   // causal band padded to 128-block boundary
  const int nv   = jmax >> 2;

  const float4* s4 = reinterpret_cast<const float4*>(Sc + (size_t)i * n);
  for (int v = tid; v < nv; v += 256)
    *reinterpret_cast<float4*>(&rowbuf[v * 4]) = s4[v];
  __syncthreads();

  float m = -3.0e38f;
  for (int j = tid; j < len; j += 256) m = fmaxf(m, rowbuf[j]);
  red[tid] = m; __syncthreads();
  for (int s = 128; s > 0; s >>= 1) { if (tid < s) red[tid] = fmaxf(red[tid], red[tid + s]); __syncthreads(); }
  m = red[0]; __syncthreads();

  float sum = 0.f;
  for (int j = tid; j < len; j += 256) sum += __expf(rowbuf[j] - m);
  red[tid] = sum; __syncthreads();
  for (int s = 128; s > 0; s >>= 1) { if (tid < s) red[tid] += red[tid + s]; __syncthreads(); }
  const float inv = 1.0f / red[0];

  __bf16* prow = P + (size_t)i * n;
  for (int v = tid; v < nv; v += 256) {
    union { __bf16 e[4]; uint2 u; } o;
#pragma unroll
    for (int c = 0; c < 4; ++c) {
      int j = v * 4 + c;
      float val = (j < len) ? __expf(rowbuf[j] - m) * inv : 0.0f;
      o.e[c] = (__bf16)val;
    }
    *reinterpret_cast<uint2*>(&prow[v * 4]) = o.u;
  }
}

// ---------------- host-side orchestration ----------------
extern "C" void kernel_launch(void* const* d_in, const int* in_sizes, int n_in,
                              void* d_out, int out_size, void* d_ws, size_t ws_size,
                              hipStream_t stream) {
  const float* x  = (const float*)d_in[0];
  const float* Wq = (const float*)d_in[1];
  const float* Wk = (const float*)d_in[2];
  const float* Wv = (const float*)d_in[3];
  float* out = (float*)d_out;
  char* ws = (char*)d_ws;

  const size_t MB = 1024ull * 1024ull;
  __bf16* xb  = (__bf16*)(ws);              // 16 MB : x bf16 [S][D]
  __bf16* Wqt = (__bf16*)(ws + 16 * MB);    // 2 MB each : W^T bf16 [Dout][Din]
  __bf16* Wkt = (__bf16*)(ws + 18 * MB);
  __bf16* Wvt = (__bf16*)(ws + 20 * MB);
  __bf16* Qb  = (__bf16*)(ws + 22 * MB);    // 16 MB : Q [S][D]
  __bf16* Kb  = (__bf16*)(ws + 38 * MB);    // 16 MB : K [S][D]  (= [N][K] operand)
  __bf16* Vt  = (__bf16*)(ws + 54 * MB);    // 16 MB : V^T [D][S] (= [N][K] operand)
  float*  Sc  = (float*) (ws + 70 * MB);    // 256 MB fp32 scores
  __bf16* P   = (__bf16*)(ws + 326 * MB);   // 128 MB bf16 probabilities

  // conversions / weight transposes
  int n4 = (S_SEQ * D_DIM) / 4;
  cvt_f32_bf16_v4<<<(n4 + 255) / 256, 256, 0, stream>>>((const float4*)x, (uint2*)xb, n4);
  dim3 gT(D_DIM / 32, D_DIM / 32);
  cvt_transpose_f32_bf16<<<gT, 256, 0, stream>>>(Wq, Wqt, D_DIM, D_DIM);
  cvt_transpose_f32_bf16<<<gT, 256, 0, stream>>>(Wk, Wkt, D_DIM, D_DIM);
  cvt_transpose_f32_bf16<<<gT, 256, 0, stream>>>(Wv, Wvt, D_DIM, D_DIM);

  dim3 blk(256);
  // Q/K projections -> [S][D] bf16 ; V projection -> V^T [D][S] bf16 (vector epilogue)
  dim3 gQKV(D_DIM / BN, S_SEQ / BM);
  gemm_wmma_bf16<true, false, false, false><<<gQKV, blk, 0, stream>>>(xb, Wqt, Qb, S_SEQ, D_DIM, D_DIM, 1.0f);
  gemm_wmma_bf16<true, false, false, false><<<gQKV, blk, 0, stream>>>(xb, Wkt, Kb, S_SEQ, D_DIM, D_DIM, 1.0f);
  gemm_wmma_bf16<true, true,  false, false><<<gQKV, blk, 0, stream>>>(xb, Wvt, Vt, S_SEQ, D_DIM, D_DIM, 1.0f);

  // scores = (Q @ K^T) / 32, fp32, lower-triangular blocks only
  dim3 gS(S_SEQ / BN, S_SEQ / BM);
  gemm_wmma_bf16<false, false, true, false><<<gS, blk, 0, stream>>>(Qb, Kb, Sc, S_SEQ, S_SEQ, D_DIM, 1.0f / 32.0f);

  // causal softmax -> P (bf16)
  softmax_causal<<<S_SEQ, 256, 0, stream>>>(Sc, P, S_SEQ);

  // out = P @ (V^T)^T, k-loop limited to the causal band
  dim3 gO(D_DIM / BN, S_SEQ / BM);
  gemm_wmma_bf16<false, false, false, true><<<gO, blk, 0, stream>>>(P, Vt, out, S_SEQ, D_DIM, S_SEQ, 1.0f);
}
